// Encoder_58720792871424
// MI455X (gfx1250) — compile-verified, matrix operand order checked by reading, not run on hardware
//
#include <hip/hip_runtime.h>
#include <hip/hip_bf16.h>
#include <math.h>

// ---------------------------------------------------------------------------
// Bidirectional skip-LSTM encoder for gfx1250 (MI455X), f16 WMMA everywhere.
//   B=32, T=1024, E=256, H=256, 4H=1024, VOCAB=50000
// Pipeline:
//   embed_cvt : embedding table f32 -> f16 (once)
//   prep      : swizzle weights into WMMA B-fragment layout (f16), fuse biases
//   gx        : embedding gather via async global->LDS (f16) +
//               [32768,256]x[256,1024] WMMA GEMM -> gx in C-fragment layout
//   lstm      : 2 persistent workgroups (one per direction), 32 waves each;
//               per step: vectorized ctx skip-sum -> LDS, WMMA gate GEMM with
//               gx fragments as C-init, LSTM cell in registers, h -> history
//               (agent-scope release/acquire fences for cross-CU visibility)
//   proj      : [32,512]x[512,256] WMMA projections for final h and c
//   mask      : token==0 mask
// ---------------------------------------------------------------------------

typedef _Float16 v16h __attribute__((ext_vector_type(16)));
typedef float    v8f  __attribute__((ext_vector_type(8)));

union Frag16 { uint4 u[2]; v16h v; };
union Half8  { uint4 u;    _Float16 h[8]; };

#define NB      32
#define NT_     1024
#define NE      256
#define NH      256
#define NG4     1024
#define LDSA_STRIDE 264   // 256 + 8 halfs pad -> bank-conflict-free ds reads

// workspace layout (_Float16 units unless noted)
#define GX_HALFS_PER_DIR   33554432ull  // T * 2 part * 64 nt * 32 lane * 8
#define HBUF_HALFS_PER_DIR  8388608ull  // T*B*H
#define WS_GX    0ull
#define WS_HBUF  67108864ull
#define WS_WIH   83886080ull
#define WS_WHH   84410368ull
#define WS_WHP   84934656ull
#define WS_WCP   85065728ull
#define WS_EMBH  85196800ull            // 50000*256 f16 embedding table
#define WS_BSUM  97996800ull            // float* from here (2048 floats + cT)

#define OUT_HPROJ 16777216ull           // B*T*2H
#define OUT_CPROJ 16785408ull
#define OUT_MASK  16793600ull

// -------------------------------------------------------------------------
// Embedding table f32 -> f16 (one-shot)
// -------------------------------------------------------------------------
__global__ void embed_cvt_kernel(const float* __restrict__ embed,
                                 _Float16* __restrict__ embh)
{
    size_t gid = (size_t)blockIdx.x * blockDim.x + threadIdx.x;
    if (gid < 12800000ull) embh[gid] = (_Float16)embed[gid];
}

// -------------------------------------------------------------------------
// Weight swizzle into fragment-major B layout + bias fuse.
// B[k][n] = W[n][k].  Fragment (kt,nt): 32 lanes x 16 halfs contiguous.
// lane<16 holds k in {0..7,16..23}+kt*32, lane>=16 k+8.
// -------------------------------------------------------------------------
__global__ void prep_kernel(const float* __restrict__ Wih_f, const float* __restrict__ Whh_f,
                            const float* __restrict__ bih_f, const float* __restrict__ bhh_f,
                            const float* __restrict__ Wih_b, const float* __restrict__ Whh_b,
                            const float* __restrict__ bih_b, const float* __restrict__ bhh_b,
                            const float* __restrict__ Whp,   const float* __restrict__ Wcp,
                            _Float16* __restrict__ wsh, float* __restrict__ bsum)
{
    int z   = blockIdx.y;
    int gid = blockIdx.x * blockDim.x + threadIdx.x;

    if (z == 6) {  // fused biases: bsum[dir][i] = b_ih[i] + b_hh[i]
        if (gid < 2048) {
            int d = gid >> 10, i = gid & 1023;
            const float* bi = d ? bih_b : bih_f;
            const float* bh = d ? bhh_b : bhh_f;
            bsum[d * 1024 + i] = bi[i] + bh[i];
        }
        return;
    }

    const float* src = nullptr; _Float16* dst = nullptr; int N = 0, K = 0;
    switch (z) {
        case 0: src = Wih_f; dst = wsh + WS_WIH;          N = NG4; K = NE;  break;
        case 1: src = Wih_b; dst = wsh + WS_WIH + 262144; N = NG4; K = NE;  break;
        case 2: src = Whh_f; dst = wsh + WS_WHH;          N = NG4; K = NH;  break;
        case 3: src = Whh_b; dst = wsh + WS_WHH + 262144; N = NG4; K = NH;  break;
        case 4: src = Whp;   dst = wsh + WS_WHP;          N = NH;  K = 512; break;
        case 5: src = Wcp;   dst = wsh + WS_WCP;          N = NH;  K = 512; break;
        default: return;
    }
    int NTn = N >> 4;
    int frags = (K >> 5) * NTn;
    if (gid >= frags * 32) return;
    int fid  = gid >> 5;
    int lane = gid & 31;
    int kt = fid / NTn, nt = fid % NTn;
    int n  = nt * 16 + (lane & 15);
    int kb = (lane < 16) ? 0 : 8;
    _Float16* dp = dst + ((size_t)fid * 32 + lane) * 16;
    #pragma unroll
    for (int h = 0; h < 16; ++h) {
        int kl = (h & 7) + kb + ((h >> 3) << 4);
        dp[h] = (_Float16)src[(size_t)n * K + kt * 32 + kl];
    }
}

// -------------------------------------------------------------------------
// gx = embed[tokens] @ W_ih^T + (b_ih+b_hh).  Block tile 64(M) x 128(N).
// A tile staged with gfx1250 async global->LDS b128 copies (f16 table).
// Output stored in C-fragment-major layout: [t][part][nt][lane][8] halfs.
// -------------------------------------------------------------------------
__global__ __launch_bounds__(256)
void gx_kernel(const int* __restrict__ tokens, const int* __restrict__ rev_tokens,
               const _Float16* __restrict__ embh, const _Float16* __restrict__ wsh_wih,
               const float* __restrict__ bsum, _Float16* __restrict__ gx)
{
    __shared__ alignas(16) _Float16 As[64 * LDSA_STRIDE];

    int dir = blockIdx.z;
    const int*      tok = dir ? rev_tokens : tokens;
    const _Float16* wB  = wsh_wih + (size_t)dir * 262144;
    const float*    bs  = bsum + dir * 1024;
    _Float16*       gxd = gx + (size_t)dir * GX_HALFS_PER_DIR;

    int tid = threadIdx.x;
    int rowBase = blockIdx.x * 64;
    int w = tid >> 5, lane = tid & 31;

    // ---- stage A tile (64 rows x 256 halfs): async DMA, one row per issue
    #pragma unroll
    for (int rr = 0; rr < 8; ++rr) {
        int rl  = w * 8 + rr;             // local row 0..63
        int row = rowBase + rl;           // global row = t*32 + b
        int t = row >> 5, b = row & 31;
        int token = tok[b * NT_ + t];
        unsigned long long ga =
            (unsigned long long)(uintptr_t)(embh + (size_t)token * NE + lane * 8);
        unsigned int la =
            (unsigned int)(uintptr_t)(As + rl * LDSA_STRIDE + lane * 8);
        asm volatile("global_load_async_to_lds_b128 %0, %1, off"
                     :: "v"(la), "v"(ga) : "memory");
    }
    asm volatile("s_wait_asynccnt 0" ::: "memory");
    __syncthreads();

    // ---- WMMA: wave w -> mtl=w&3 (M-tile), ng=w>>2 (4 N-tiles) ----------
    int mtl = w & 3;
    int ng  = w >> 2;
    int ntBase = blockIdx.y * 8;

    v8f acc[4];
    #pragma unroll
    for (int i = 0; i < 4; ++i) {
        float bv = bs[(ntBase + ng * 4 + i) * 16 + (lane & 15)];
        #pragma unroll
        for (int r = 0; r < 8; ++r) acc[i][r] = bv;   // C init = fused bias
    }

    for (int kt = 0; kt < 8; ++kt) {
        Frag16 a;
        const _Float16* ap = As + (mtl * 16 + (lane & 15)) * LDSA_STRIDE
                               + kt * 32 + ((lane < 16) ? 0 : 8);
        a.u[0] = *(const uint4*)ap;
        a.u[1] = *(const uint4*)(ap + 16);
        #pragma unroll
        for (int i = 0; i < 4; ++i) {
            int ntg = ntBase + ng * 4 + i;
            Frag16 b;
            const _Float16* bp = wB + ((size_t)(kt * 64 + ntg) * 32 + lane) * 16;
            b.u[0] = *(const uint4*)bp;
            b.u[1] = *(const uint4*)(bp + 8);
            acc[i] = __builtin_amdgcn_wmma_f32_16x16x32_f16(
                         false, a.v, false, b.v, (short)0, acc[i], false, false);
        }
    }

    // ---- store gx in C-fragment-major layout (one b128 per tile) --------
    int row0 = rowBase + mtl * 16;
    int tt   = row0 >> 5;                 // timestep of this 16-row tile
    int part = (row0 >> 4) & 1;           // b-half (0: b0..15, 1: b16..31)
    #pragma unroll
    for (int i = 0; i < 4; ++i) {
        int ntg = ntBase + ng * 4 + i;
        Half8 pk;
        #pragma unroll
        for (int r = 0; r < 8; ++r) pk.h[r] = (_Float16)acc[i][r];
        *(uint4*)(gxd + (((size_t)(tt * 2 + part) * 64 + ntg) * 32 + lane) * 8) = pk.u;
    }
}

// -------------------------------------------------------------------------
// Sequential skip-LSTM scan.  grid=2 (dir), block=1024 (32 waves).
// wave w -> mt=w&1 (b-half), g=w>>1 (16 h-cols); owns tiles {g,g+16,g+32,g+48}
// = gates i,f,g,o for its columns -> full LSTM cell + c-state in registers.
// -------------------------------------------------------------------------
__global__ __launch_bounds__(1024)
void lstm_kernel(const int* __restrict__ tokens, const _Float16* __restrict__ gx,
                 const _Float16* __restrict__ whh, _Float16* __restrict__ hbuf,
                 float* __restrict__ ctws, float* __restrict__ out)
{
    __shared__ alignas(16) _Float16 ctxs[32 * LDSA_STRIDE];

    int dir = blockIdx.x;
    const _Float16* gxd = gx  + (size_t)dir * GX_HALFS_PER_DIR;
    const _Float16* wB  = whh + (size_t)dir * 262144;
    _Float16*       hb  = hbuf + (size_t)dir * HBUF_HALFS_PER_DIR;

    int tid = threadIdx.x;
    int w = tid >> 5, lane = tid & 31;
    int mt = w & 1;
    int g  = w >> 1;                       // 0..15
    int col = g * 16 + (lane & 15);        // h column owned by this lane
    int rowHalf = (lane < 16) ? 0 : 8;

    int base = tid * 8;                    // ctx chunk: fixed b, 8 h's
    int cb   = base >> 8;
    int ch0  = base & 255;

    float c[8];
    #pragma unroll
    for (int r = 0; r < 8; ++r) c[r] = 0.f;

    for (int t = 0; t < NT_; ++t) {
        // make prior-step h stores (other CU) visible to our loads
        __builtin_amdgcn_fence(__ATOMIC_ACQUIRE, "agent");

        // ---- ctx_t = sum_k h[t - 2^k] -> LDS, vectorized b128 ----------
        int K = 0; { int tt = t; while (tt) { ++K; tt >>= 1; } }
        float s[8];
        #pragma unroll
        for (int e = 0; e < 8; ++e) s[e] = 0.f;
        for (int k = 0; k < K; ++k) {
            int tj = t - (1 << k);
            Half8 d;
            d.u = *(const uint4*)(hb + (size_t)tj * 8192 + base);
            #pragma unroll
            for (int e = 0; e < 8; ++e) s[e] += (float)d.h[e];
        }
        Half8 o;
        #pragma unroll
        for (int e = 0; e < 8; ++e) o.h[e] = (_Float16)s[e];
        *(uint4*)(ctxs + cb * LDSA_STRIDE + ch0) = o.u;
        __syncthreads();

        // ---- gates = gx_t + ctx @ W_hh^T  (C init = gx fragments) ------
        v8f acc[4];
        #pragma unroll
        for (int j = 0; j < 4; ++j) {      // j = gate (i,f,g,o); nt = g+16j
            int nt = g + 16 * j;
            Half8 pk;
            pk.u = *(const uint4*)(gxd + (((size_t)(t * 2 + mt) * 64 + nt) * 32 + lane) * 8);
            #pragma unroll
            for (int r = 0; r < 8; ++r) acc[j][r] = (float)pk.h[r];
        }
        for (int kt = 0; kt < 8; ++kt) {
            Frag16 a;
            const _Float16* ap = ctxs + (mt * 16 + (lane & 15)) * LDSA_STRIDE
                                      + kt * 32 + ((lane < 16) ? 0 : 8);
            a.u[0] = *(const uint4*)ap;
            a.u[1] = *(const uint4*)(ap + 16);
            #pragma unroll
            for (int j = 0; j < 4; ++j) {
                int nt = g + 16 * j;
                Frag16 b;
                const _Float16* bp = wB + ((size_t)(kt * 64 + nt) * 32 + lane) * 16;
                b.u[0] = *(const uint4*)bp;
                b.u[1] = *(const uint4*)(bp + 8);
                acc[j] = __builtin_amdgcn_wmma_f32_16x16x32_f16(
                             false, a.v, false, b.v, (short)0, acc[j], false, false);
            }
        }

        // ---- LSTM cell, all in registers -------------------------------
        #pragma unroll
        for (int r = 0; r < 8; ++r) {
            int b = mt * 16 + rowHalf + r;
            float iv = acc[0][r], fv = acc[1][r], gv = acc[2][r], ov = acc[3][r];
            float si = 1.f / (1.f + expf(-iv));
            float sf = 1.f / (1.f + expf(-fv));
            float so = 1.f / (1.f + expf(-ov));
            float cn = sf * c[r] + si * tanhf(gv);
            float hv = so * tanhf(cn);
            c[r] = cn;
            hb[(size_t)t * 8192 + b * 256 + col] = (_Float16)hv;
            float inv = (tokens[b * NT_ + t] == 0) ? 0.f : 1.f;
            out[(size_t)b * 524288 + (size_t)t * 512 + dir * 256 + col] = hv * inv;
            if (t == NT_ - 1) ctws[dir * 8192 + b * 256 + col] = cn;
        }
        // publish this step's h to both CUs before anyone starts step t+1
        __builtin_amdgcn_fence(__ATOMIC_RELEASE, "agent");
        __syncthreads();
    }
}

// -------------------------------------------------------------------------
// Final projections: concat(hT_f,hT_b)[32,512] @ Wproj^T + b (h and c).
// -------------------------------------------------------------------------
__global__ __launch_bounds__(1024)
void proj_kernel(const _Float16* __restrict__ hbuf, const float* __restrict__ ctws,
                 const _Float16* __restrict__ whp, const _Float16* __restrict__ wcp,
                 const float* __restrict__ bh, const float* __restrict__ bc,
                 float* __restrict__ out)
{
    int tid = threadIdx.x;
    int w = tid >> 5, lane = tid & 31;
    int mt = w & 1, nt = w >> 1;           // nt 0..15
    int col = nt * 16 + (lane & 15);
    int rowA = mt * 16 + (lane & 15);
    int rowHalf = (lane < 16) ? 0 : 8;
    int kb = (lane < 16) ? 0 : 8;

    v8f accH, accC;
    #pragma unroll
    for (int r = 0; r < 8; ++r) { accH[r] = bh[col]; accC[r] = bc[col]; }

    const _Float16* hbF = hbuf;
    const _Float16* hbB = hbuf + HBUF_HALFS_PER_DIR;

    for (int kt = 0; kt < 16; ++kt) {
        v16h aH, aC;
        #pragma unroll
        for (int h = 0; h < 16; ++h) {
            int kl = (h & 7) + kb + ((h >> 3) << 4);
            int k  = kt * 32 + kl;         // 0..511 over concat(fwd,bwd)
            int d  = (k >= 256);
            int hh = k & 255;
            const _Float16* hbd = d ? hbB : hbF;
            aH[h] = hbd[(size_t)1023 * 8192 + rowA * 256 + hh];
            aC[h] = (_Float16)ctws[d * 8192 + rowA * 256 + hh];
        }
        Frag16 bH, bC;
        const _Float16* bpH = whp + ((size_t)(kt * 16 + nt) * 32 + lane) * 16;
        const _Float16* bpC = wcp + ((size_t)(kt * 16 + nt) * 32 + lane) * 16;
        bH.u[0] = *(const uint4*)bpH; bH.u[1] = *(const uint4*)(bpH + 8);
        bC.u[0] = *(const uint4*)bpC; bC.u[1] = *(const uint4*)(bpC + 8);
        accH = __builtin_amdgcn_wmma_f32_16x16x32_f16(
                   false, aH, false, bH.v, (short)0, accH, false, false);
        accC = __builtin_amdgcn_wmma_f32_16x16x32_f16(
                   false, aC, false, bC.v, (short)0, accC, false, false);
    }
    #pragma unroll
    for (int r = 0; r < 8; ++r) {
        int b = mt * 16 + rowHalf + r;
        out[OUT_HPROJ + (size_t)b * 256 + col] = accH[r];
        out[OUT_CPROJ + (size_t)b * 256 + col] = accC[r];
    }
}

__global__ void mask_kernel(const int* __restrict__ tokens, float* __restrict__ out)
{
    int gid = blockIdx.x * blockDim.x + threadIdx.x;
    if (gid < NB * NT_)
        out[OUT_MASK + gid] = (tokens[gid] == 0) ? 1.f : 0.f;
}

// -------------------------------------------------------------------------
extern "C" void kernel_launch(void* const* d_in, const int* in_sizes, int n_in,
                              void* d_out, int out_size, void* d_ws, size_t ws_size,
                              hipStream_t stream)
{
    (void)in_sizes; (void)n_in; (void)out_size; (void)ws_size;

    const int*   tokens     = (const int*)d_in[0];
    const int*   rev_tokens = (const int*)d_in[1];
    const float* embed      = (const float*)d_in[2];
    const float* W_ih_f     = (const float*)d_in[3];
    const float* W_hh_f     = (const float*)d_in[4];
    const float* b_ih_f     = (const float*)d_in[5];
    const float* b_hh_f     = (const float*)d_in[6];
    const float* W_ih_b     = (const float*)d_in[7];
    const float* W_hh_b     = (const float*)d_in[8];
    const float* b_ih_b     = (const float*)d_in[9];
    const float* b_hh_b     = (const float*)d_in[10];
    const float* W_hproj    = (const float*)d_in[11];
    const float* b_hproj    = (const float*)d_in[12];
    const float* W_cproj    = (const float*)d_in[13];
    const float* b_cproj    = (const float*)d_in[14];
    float* out = (float*)d_out;

    _Float16* wsh  = (_Float16*)d_ws;
    _Float16* gx   = wsh + WS_GX;
    _Float16* hbuf = wsh + WS_HBUF;
    _Float16* wih  = wsh + WS_WIH;
    _Float16* whh  = wsh + WS_WHH;
    _Float16* whp  = wsh + WS_WHP;
    _Float16* wcp  = wsh + WS_WCP;
    _Float16* embh = wsh + WS_EMBH;
    float*    bsum = (float*)(wsh + WS_BSUM);
    float*    ctws = bsum + 2048;

    // 1) embedding table -> f16
    embed_cvt_kernel<<<dim3(50000, 1, 1), 256, 0, stream>>>(embed, embh);

    // 2) weight swizzle + bias fuse
    prep_kernel<<<dim3(64, 7, 1), 256, 0, stream>>>(
        W_ih_f, W_hh_f, b_ih_f, b_hh_f, W_ih_b, W_hh_b, b_ih_b, b_hh_b,
        W_hproj, W_cproj, wsh, bsum);

    // 3) mask output (independent)
    mask_kernel<<<dim3(128, 1, 1), 256, 0, stream>>>(tokens, out);

    // 4) gx GEMM: M=32768 (512x64), N=1024 (8x128), 2 dirs
    gx_kernel<<<dim3(512, 8, 2), 256, 0, stream>>>(
        tokens, rev_tokens, embh, wih, bsum, gx);

    // 5) sequential scan: one persistent workgroup per direction
    lstm_kernel<<<dim3(2, 1, 1), 1024, 0, stream>>>(
        tokens, gx, whh, hbuf, ctws, out);

    // 6) final projections
    proj_kernel<<<dim3(1, 1, 1), 1024, 0, stream>>>(
        hbuf, ctws, whp, wcp, b_hproj, b_cproj, out);
}
